// GptOssExpertsParallel_64888365908031
// MI455X (gfx1250) — matrix operand order repaired and checked by reading, not compiled, not run on hardware
//
#include <hip/hip_runtime.h>
#include <hip/hip_bf16.h>

// ---------------------------------------------------------------------------
// GPT-OSS MoE experts layer for gfx1250 (MI455X), bf16 WMMA path.
//  - weights converted fp32->bf16 AND transposed once (K-contiguous per column)
//  - grouped GEMM: block tile 128x128, 8 waves, wave tile 32x64
//  - double-buffered LDS fed by global_load_async_to_lds_b128 (ASYNCcnt),
//    one barrier per K-step, copies overlap WMMA
// ---------------------------------------------------------------------------

typedef __attribute__((ext_vector_type(16))) __bf16 v16bf;
typedef __attribute__((ext_vector_type(8)))  float  v8f;

#define NUM_EXPERTS 8
#define HIDDEN      1024
#define INTER       1024
#define M_TOKENS    2048
#define TOP_K       2
#define T_SLOTS     (M_TOKENS * TOP_K)          // 4096
#define SWIGLU_ALPHA 1.702f
#define LIMIT        7.0f

// GEMM tiling
#define TM    128
#define TN    128
#define KSTEP 32
#define LDA   40          // ushorts per LDS row (32 + 8 pad); 80B rows keep 16B align
#define LDB   40
#define MAX_MBLK 32       // ceil(4096/128): worst case all slots on one expert

__device__ __forceinline__ unsigned short f2bf(float f) {
    union { float f; unsigned u; } x; x.f = f;
    unsigned r = x.u + 0x7FFFu + ((x.u >> 16) & 1u);   // round-to-nearest-even
    return (unsigned short)(r >> 16);
}

__device__ __forceinline__ void async_b128(unsigned lds_addr, const void* gaddr) {
    asm volatile("global_load_async_to_lds_b128 %0, %1, off"
                 :: "v"(lds_addr), "v"(gaddr) : "memory");
}
__device__ __forceinline__ void wait_async0() {
    asm volatile("s_wait_asynccnt 0x0" ::: "memory");
}

// ---------------------------------------------------------------------------
// Dispatch kernels (int atomics only; output remains bitwise deterministic:
// within-expert slot order never changes any per-slot value, and the combine
// gathers by recorded position)
// ---------------------------------------------------------------------------
__global__ void k_init(int* counts) {
    if (threadIdx.x < NUM_EXPERTS) counts[threadIdx.x] = 0;
}

__global__ void k_count(const int* __restrict__ ridx, int* __restrict__ counts) {
    int t = blockIdx.x * blockDim.x + threadIdx.x;
    if (t < T_SLOTS) atomicAdd(&counts[ridx[t]], 1);
}

__global__ void k_scan(const int* __restrict__ counts, int* __restrict__ offs,
                       int* __restrict__ cursor) {
    if (threadIdx.x == 0) {
        int s = 0;
        for (int e = 0; e < NUM_EXPERTS; ++e) { offs[e] = s; cursor[e] = s; s += counts[e]; }
    }
}

__global__ void k_place(const int* __restrict__ ridx, int* __restrict__ cursor,
                        int* __restrict__ slot_token, int* __restrict__ pos_of_slot) {
    int t = blockIdx.x * blockDim.x + threadIdx.x;
    if (t < T_SLOTS) {
        int e = ridx[t];
        int p = atomicAdd(&cursor[e], 1);
        slot_token[p]  = t >> 1;     // token id
        pos_of_slot[t] = p;          // where this (token,k) slot landed
    }
}

// ---------------------------------------------------------------------------
// Weight transpose + fp32->bf16:  src [E][1024][N] f32  ->  dst [E][N][1024] bf16
// 32(K) x 64(N) tiles via LDS; coalesced reads and b128 writes.
// ---------------------------------------------------------------------------
__global__ __launch_bounds__(256)
void k_cvt_t(const float* __restrict__ src, unsigned short* __restrict__ dst, int N) {
    __shared__ __align__(16) unsigned short t_[64 * LDA];   // [n][k] padded
    const int tilesK = HIDDEN / 32, tilesN = N / 64;
    const int bid = blockIdx.x;
    const int e   = bid / (tilesK * tilesN);
    const int rem = bid % (tilesK * tilesN);
    const int k0  = (rem / tilesN) * 32;
    const int n0  = (rem % tilesN) * 64;

    for (int i = threadIdx.x; i < 32 * 64; i += 256) {
        int kk = i >> 6, nn = i & 63;                       // n contiguous -> coalesced
        float v = src[((size_t)e * HIDDEN + (k0 + kk)) * N + n0 + nn];
        t_[nn * LDA + kk] = f2bf(v);
    }
    __syncthreads();
    int nn = threadIdx.x >> 2, ko = (threadIdx.x & 3) * 8;
    uint4 v = *(const uint4*)(&t_[nn * LDA + ko]);
    *(uint4*)(dst + ((size_t)e * N + n0 + nn) * HIDDEN + k0 + ko) = v;
}

// ---------------------------------------------------------------------------
// Activation gather rows -> bf16  (rows indexed by sorted position)
// ---------------------------------------------------------------------------
__global__ void k_gather(const float* __restrict__ x, const int* __restrict__ slot_token,
                         unsigned short* __restrict__ Xg) {
    int p = blockIdx.x;
    int t = slot_token[p];
    for (int h = threadIdx.x; h < HIDDEN; h += blockDim.x)
        Xg[(size_t)p * HIDDEN + h] = f2bf(x[(size_t)t * HIDDEN + h]);
}

// ---------------------------------------------------------------------------
// Grouped GEMM, bf16 WMMA 16x16x32, fp32 accumulate.
//   A : [4096][1024] bf16, rows = sorted positions
//   Wt: [E][N][1024]  bf16, K contiguous (pre-transposed)
// Block 128x128, 8 waves (4 M-sub x 2 N-sub), wave tile 32x64 -> 8 accumulators.
// A rows past the group end are clamped to the last valid row (finite garbage,
// never stored by the bounds-checked epilogue).
// ---------------------------------------------------------------------------
template<bool SWIGLU>
__global__ __launch_bounds__(256)
void k_grouped_gemm(const unsigned short* __restrict__ A,
                    const unsigned short* __restrict__ Wt,
                    const int* __restrict__ offs, const int* __restrict__ cnts,
                    const float* __restrict__ bias,
                    unsigned short* __restrict__ outInter,
                    float* __restrict__ outDown,
                    int N)
{
    const int nblocks = N / TN;
    const int bid  = blockIdx.x;
    const int e    = bid / (MAX_MBLK * nblocks);
    const int rem  = bid % (MAX_MBLK * nblocks);
    const int mblk = rem / nblocks;
    const int nblk = rem % nblocks;

    const int cntE = cnts[e];
    if (mblk * TM >= cntE) return;
    const int offE = offs[e];

    __shared__ __align__(16) unsigned short Al[2][TM * LDA];
    __shared__ __align__(16) unsigned short Bl[2][TN * LDB];

    const int tid  = threadIdx.x;
    const int lane = tid & 31;
    const int w    = tid >> 5;              // wave 0..7
    const int wm   = (w & 3) * 32;          // wave M sub-base
    const int wn   = (w >> 2) * 64;         // wave N sub-base
    const int lcol  = lane & 15;
    const int lhalf = lane >> 4;

    // staging: 2 b128 chunks per thread per matrix; chunk c -> row=c>>2, koff=(c&3)*8
    const int r0 = tid >> 2,         ko0 = (tid & 3) * 8;
    const int r1 = (tid + 256) >> 2, ko1 = ko0;            // rows 64..127
    const int a0row = (mblk * TM + r0 < cntE) ? (mblk * TM + r0) : (cntE - 1);
    const int a1row = (mblk * TM + r1 < cntE) ? (mblk * TM + r1) : (cntE - 1);

    const unsigned short* gA0 = A + ((size_t)offE + a0row) * HIDDEN + ko0;
    const unsigned short* gA1 = A + ((size_t)offE + a1row) * HIDDEN + ko1;
    const unsigned short* gB0 = Wt + ((size_t)e * N + nblk * TN + r0) * HIDDEN + ko0;
    const unsigned short* gB1 = Wt + ((size_t)e * N + nblk * TN + r1) * HIDDEN + ko1;

    const unsigned lA0[2] = { (unsigned)(uintptr_t)&Al[0][r0 * LDA + ko0],
                              (unsigned)(uintptr_t)&Al[1][r0 * LDA + ko0] };
    const unsigned lA1[2] = { (unsigned)(uintptr_t)&Al[0][r1 * LDA + ko1],
                              (unsigned)(uintptr_t)&Al[1][r1 * LDA + ko1] };
    const unsigned lB0[2] = { (unsigned)(uintptr_t)&Bl[0][r0 * LDB + ko0],
                              (unsigned)(uintptr_t)&Bl[1][r0 * LDB + ko0] };
    const unsigned lB1[2] = { (unsigned)(uintptr_t)&Bl[0][r1 * LDB + ko1],
                              (unsigned)(uintptr_t)&Bl[1][r1 * LDB + ko1] };

    auto issue = [&](int buf, int k0) {
        async_b128(lA0[buf], gA0 + k0);
        async_b128(lA1[buf], gA1 + k0);
        async_b128(lB0[buf], gB0 + k0);
        async_b128(lB1[buf], gB1 + k0);
    };

    v8f acc0[4], acc1[4];
#pragma unroll
    for (int t = 0; t < 4; ++t) {
        acc0[t] = (v8f){0.f,0.f,0.f,0.f,0.f,0.f,0.f,0.f};
        acc1[t] = (v8f){0.f,0.f,0.f,0.f,0.f,0.f,0.f,0.f};
    }

    issue(0, 0);
    const int NSTEP = HIDDEN / KSTEP;                     // 32
    for (int ks = 0; ks < NSTEP; ++ks) {
        const int cur = ks & 1;
        wait_async0();            // my async copies into buf[cur] are done
        __syncthreads();          // everyone's are done; prev reads of buf[!cur] done
        if (ks + 1 < NSTEP) issue(cur ^ 1, (ks + 1) * KSTEP);   // overlap WMMA below

        union U { uint4 q[2]; v16bf v; };
        U a0, a1;
        const unsigned short* pa0 = &Al[cur][(wm + lcol) * LDA + lhalf * 8];
        a0.q[0] = *(const uint4*)(pa0);
        a0.q[1] = *(const uint4*)(pa0 + 16);
        const unsigned short* pa1 = pa0 + 16 * LDA;
        a1.q[0] = *(const uint4*)(pa1);
        a1.q[1] = *(const uint4*)(pa1 + 16);
#pragma unroll
        for (int t = 0; t < 4; ++t) {
            U b;
            const unsigned short* pb = &Bl[cur][(wn + 16 * t + lcol) * LDB + lhalf * 8];
            b.q[0] = *(const uint4*)(pb);
            b.q[1] = *(const uint4*)(pb + 16);
            acc0[t] = __builtin_amdgcn_wmma_f32_16x16x32_bf16(
                false, a0.v, false, b.v, (short)0, acc0[t], false, false);
            acc1[t] = __builtin_amdgcn_wmma_f32_16x16x32_bf16(
                false, a1.v, false, b.v, (short)0, acc1[t], false, false);
        }
    }

    // ---- epilogue ----
#pragma unroll
    for (int i = 0; i < 2; ++i) {
#pragma unroll
        for (int t = 0; t < 4; ++t) {
            const int Ncol = nblk * TN + wn + 16 * t + lcol;
            float badd = 0.f;
            if (SWIGLU) badd = bias[e * (2 * INTER) + Ncol];
#pragma unroll
            for (int r = 0; r < 8; ++r) {
                const int mrow = mblk * TM + wm + 16 * i + r + 8 * lhalf;
                float c = (i == 0 ? acc0[t][r] : acc1[t][r]) + badd;
                if (SWIGLU) {
                    float other = __shfl_xor(c, 1, 32);   // pair even(gate)/odd(up)
                    if ((lcol & 1) == 0) {
                        float gate = fminf(c, LIMIT);
                        float up   = fminf(fmaxf(other, -LIMIT), LIMIT);
                        float glu  = gate / (1.f + __expf(-gate * SWIGLU_ALPHA));
                        float val  = (up + 1.f) * glu;
                        if (mrow < cntE)
                            outInter[(size_t)(offE + mrow) * INTER + (Ncol >> 1)] = f2bf(val);
                    }
                } else {
                    if (mrow < cntE)
                        outDown[(size_t)(offE + mrow) * HIDDEN + Ncol] = c;
                }
            }
        }
    }
}

// ---------------------------------------------------------------------------
// Combine: gather both slots of each token (deterministic, no fp atomics)
// ---------------------------------------------------------------------------
__global__ void k_combine(const float* __restrict__ down_out,
                          const int* __restrict__ ridx,
                          const float* __restrict__ rweights,
                          const float* __restrict__ dbias,
                          const int* __restrict__ pos_of_slot,
                          float* __restrict__ out)
{
    int m  = blockIdx.x;
    int e0 = ridx[2 * m],     e1 = ridx[2 * m + 1];
    float w0 = rweights[m * NUM_EXPERTS + e0];
    float w1 = rweights[m * NUM_EXPERTS + e1];
    int p0 = pos_of_slot[2 * m], p1 = pos_of_slot[2 * m + 1];
    for (int h = threadIdx.x; h < HIDDEN; h += blockDim.x) {
        float v0 = down_out[(size_t)p0 * HIDDEN + h] + dbias[e0 * HIDDEN + h];
        float v1 = down_out[(size_t)p1 * HIDDEN + h] + dbias[e1 * HIDDEN + h];
        out[(size_t)m * HIDDEN + h] = w0 * v0 + w1 * v1;
    }
}

// ---------------------------------------------------------------------------
// Host launcher
// ---------------------------------------------------------------------------
extern "C" void kernel_launch(void* const* d_in, const int* in_sizes, int n_in,
                              void* d_out, int out_size, void* d_ws, size_t ws_size,
                              hipStream_t stream) {
    (void)in_sizes; (void)n_in; (void)out_size; (void)ws_size;

    const float* x      = (const float*)d_in[0];   // [2,1024,1024]
    const int*   ridx   = (const int*)  d_in[1];   // [2048,2]
    const float* rw     = (const float*)d_in[2];   // [2048,8]
    const float* gup    = (const float*)d_in[3];   // [8,1024,2048]
    const float* gup_b  = (const float*)d_in[4];   // [8,2048]
    const float* dwn    = (const float*)d_in[5];   // [8,1024,1024]
    const float* dwn_b  = (const float*)d_in[6];   // [8,1024]
    float* out = (float*)d_out;

    // workspace layout (16B-aligned slabs)
    char* ws = (char*)d_ws;
    int* counts       = (int*)(ws + 0);
    int* cursor       = (int*)(ws + 32);
    int* offs         = (int*)(ws + 64);
    int* slot_token   = (int*)(ws + 128);                          // 4096 ints
    int* pos_of_slot  = (int*)(ws + 128 + 16384);                  // 4096 ints
    unsigned short* Xg    = (unsigned short*)(ws + 33024);         // [4096][1024] bf16
    unsigned short* WguT  = (unsigned short*)(ws + 33024 + 8388608);           // 32MB [8][2048][1024]
    unsigned short* WdT   = (unsigned short*)(ws + 33024 + 8388608 + 33554432);// 16MB [8][1024][1024]
    unsigned short* inter = (unsigned short*)(ws + 33024 + 8388608 + 33554432 + 16777216);
    float* down_out       = (float*)(ws + 33024 + 8388608 + 33554432 + 16777216 + 8388608);

    // 1) dispatch
    k_init <<<1, 32, 0, stream>>>(counts);
    k_count<<<T_SLOTS / 256, 256, 0, stream>>>(ridx, counts);
    k_scan <<<1, 32, 0, stream>>>(counts, offs, cursor);
    k_place<<<T_SLOTS / 256, 256, 0, stream>>>(ridx, cursor, slot_token, pos_of_slot);

    // 2) weight transpose+convert, activation gather
    k_cvt_t<<<NUM_EXPERTS * (HIDDEN / 32) * (2 * INTER / 64), 256, 0, stream>>>(gup, WguT, 2 * INTER);
    k_cvt_t<<<NUM_EXPERTS * (HIDDEN / 32) * (HIDDEN / 64),    256, 0, stream>>>(dwn, WdT, HIDDEN);
    k_gather<<<T_SLOTS, 256, 0, stream>>>(x, slot_token, Xg);

    // 3) grouped GEMM1 + SwiGLU  (N = 2048)
    k_grouped_gemm<true><<<NUM_EXPERTS * MAX_MBLK * (2 * INTER / TN), 256, 0, stream>>>(
        Xg, WguT, offs, counts, gup_b, inter, nullptr, 2 * INTER);

    // 4) grouped GEMM2           (N = 1024)
    k_grouped_gemm<false><<<NUM_EXPERTS * MAX_MBLK * (HIDDEN / TN), 256, 0, stream>>>(
        inter, WdT, offs, counts, nullptr, nullptr, down_out, HIDDEN);

    // 5) combine (bias + routing weights, gather-based)
    k_combine<<<M_TOKENS, 256, 0, stream>>>(down_out, ridx, rw, dwn_b, pos_of_slot, out);
}
